// ActionFormer4_88132728914283
// MI455X (gfx1250) — matrix operand order repaired and checked by reading, not compiled, not run on hardware
//
#include <hip/hip_runtime.h>
#include <hip/hip_bf16.h>
#include <type_traits>

// ---------------------------------------------------------------------------
// Types for CDNA5 WMMA (wave32): v_wmma_f32_16x16x32_bf16
// ---------------------------------------------------------------------------
typedef __bf16 bf16;
typedef __attribute__((ext_vector_type(16))) __bf16 bfrag;   // A/B operand (8 VGPRs)
typedef __attribute__((ext_vector_type(8)))  __bf16 bf8;
typedef __attribute__((ext_vector_type(8)))  float  v8f;     // C/D operand (8 VGPRs)
typedef __attribute__((ext_vector_type(4)))  unsigned int v4u;
typedef __attribute__((ext_vector_type(4)))  int v4i;
typedef __attribute__((ext_vector_type(8)))  int v8i;

#ifndef __has_builtin
#define __has_builtin(x) 0
#endif
#if __has_builtin(__builtin_amdgcn_tensor_load_to_lds) && \
    __has_builtin(__builtin_amdgcn_s_wait_tensorcnt)
#define HAVE_TDM 1
#else
#define HAVE_TDM 0
#endif

// Compile-time probe: on the gfx1250 device pass the TDM builtins must exist,
// otherwise fail loudly so the next iteration switches to inline asm.
#if defined(__gfx1250__) && !HAVE_TDM
#error "gfx1250 device pass without __builtin_amdgcn_tensor_load_to_lds"
#endif

#define DEVI static __device__ __forceinline__

DEVI bfrag cat8(bf8 lo, bf8 hi) {
  return __builtin_shufflevector(lo, hi, 0,1,2,3,4,5,6,7,8,9,10,11,12,13,14,15);
}

DEVI v8f wmma_bf16(bfrag a, bfrag b, v8f c) {
  // (neg_a, A, neg_b, B, c_mod, C, reuse_a, reuse_b)
  return __builtin_amdgcn_wmma_f32_16x16x32_bf16(false, a, false, b, (short)0, c, false, false);
}

#if HAVE_TDM
// Issue a TDM DMA of a 2-D bf16 tile (rows x tile_k, row stride = stride_elems)
// from global memory into LDS at byte offset lds_off. Descriptor per ISA ch.8.
DEVI void tdm_load_tile_bf16(unsigned lds_off, const bf16* g, unsigned rows,
                             unsigned tile_k, unsigned long long stride_elems) {
  unsigned long long ga = (unsigned long long)(size_t)g;
  v4u g0;
  g0[0] = 1u;                                             // count=1, load, user mode
  g0[1] = lds_off;                                        // lds_addr (bytes)
  g0[2] = (unsigned)ga;                                   // global_addr[31:0]
  g0[3] = (unsigned)((ga >> 32) & 0x01ffffffULL) | (2u << 30);  // addr[56:32], type=2
  v8i g1;
  g1[0] = 1 << 16;                                        // data_size=1 (2 bytes), mask=0
  g1[1] = (int)((tile_k & 0xffffu) << 16);                // tensor_dim0 lo16 -> bits[63:48]
  g1[2] = (int)(((tile_k >> 16) & 0xffffu) | ((rows & 0xffffu) << 16));  // dim0 hi | dim1 lo
  g1[3] = (int)(((rows >> 16) & 0xffffu) | ((tile_k & 0xffffu) << 16));  // dim1 hi | tile_dim0
  g1[4] = (int)(rows & 0xffffu);                          // tile_dim1 = rows, tile_dim2 = 0
  g1[5] = (int)(stride_elems & 0xffffffffULL);            // tensor_dim0_stride lo32
  g1[6] = (int)((stride_elems >> 32) & 0xffffULL);        // stride hi16 | dim1_stride lo16=0
  g1[7] = 0;
  v4i z4 = {0, 0, 0, 0};
#if __clang_major__ >= 23
  v8i z8 = {0, 0, 0, 0, 0, 0, 0, 0};
  __builtin_amdgcn_tensor_load_to_lds(g0, g1, z4, z4, z8, 0);
#else
  __builtin_amdgcn_tensor_load_to_lds(g0, g1, z4, z4, 0);
#endif
}
#endif

// ---------------------------------------------------------------------------
// Fragment loaders (wave32); A and B share contiguous K order k = 16*half + s, not the ISA table order.
// A (16x32, MxK, 16-bit): lane m = lane%16; half 0 holds K0..15, half 1 K16..31.
// B (32x16, KxN, 16-bit): lane n = lane%16; half 0 holds K0..15, half 1 K16..31.
// C/D (16x16 f32): lane n = lane%16; VGPR r: half 0 -> M=r, half 1 -> M=8+r.
// ---------------------------------------------------------------------------

DEVI bfrag load_a_g(const float* __restrict__ A, int lda, int m0, int k0, int M, int K) {
  int lane = threadIdx.x & 31, half = lane >> 4, l16 = lane & 15;
  int m = m0 + l16;
  bfrag f;
#pragma unroll
  for (int v = 0; v < 8; ++v) {
    int k = k0 + half * 16 + v * 2;
    long base = (long)m * lda + k;
    f[2*v]   = (m < M && k     < K) ? (bf16)A[base]     : (bf16)0.f;
    f[2*v+1] = (m < M && k + 1 < K) ? (bf16)A[base + 1] : (bf16)0.f;
  }
  return f;
}

// A fragment from a bf16 tile in LDS, row-major with leading dim ld (fully valid)
DEVI bfrag load_a_lds_ld(const bf16* As, int ld) {
  int lane = threadIdx.x & 31, half = lane >> 4, l16 = lane & 15;
  const bf16* p = As + l16 * ld + half * 16;
  bf8 lo = *(const bf8*)p;          // K = half*16 .. +7
  bf8 hi = *(const bf8*)(p + 8);    // K = half*16 + 8 .. +15
  return cat8(lo, hi);
}

// B fragment, f32 source laid out N x K with K contiguous: B[k][n] = W[n*ldk + k]
// NT = stream with non-temporal hint (single-use weights; keep L2 for W_hh).
template<bool NT>
DEVI bfrag load_b_kc_f32(const float* __restrict__ W, int ldk, int n0, int k0, int N, int K) {
  int lane = threadIdx.x & 31, half = lane >> 4, l16 = lane & 15;
  int n = n0 + l16;
  bfrag f;
#pragma unroll
  for (int v = 0; v < 16; ++v) {
    int k = k0 + half * 16 + v;
    float x = 0.f;
    if (n < N && k < K) {
      const float* p = W + (long)n * ldk + k;
      x = NT ? __builtin_nontemporal_load(p) : *p;
    }
    f[v] = (bf16)x;
  }
  return f;
}

// B fragment, bf16 source laid out N x K with K contiguous (temporal: L2-resident W_hh)
DEVI bfrag load_b_kc_bf16(const bf16* __restrict__ W, int ldk, int n0, int k0, int N, int K) {
  int lane = threadIdx.x & 31, half = lane >> 4, l16 = lane & 15;
  int n = n0 + l16;
  int k = k0 + half * 16;
  if (n < N && k + 15 < K) {        // fast path: one 32B contiguous load per lane
    const bf16* p = W + (long)n * ldk + k;
    bf8 lo = *(const bf8*)p;
    bf8 hi = *(const bf8*)(p + 8);
    return cat8(lo, hi);
  }
  bfrag f;
#pragma unroll
  for (int v = 0; v < 16; ++v) {
    int kk = k + v;
    f[v] = (n < N && kk < K) ? W[(long)n * ldk + kk] : (bf16)0.f;
  }
  return f;
}

// B fragment, f32 source laid out K x N with N contiguous: B[k][n] = W[k*ldn + n]
DEVI bfrag load_b_nc_f32(const float* __restrict__ W, int ldn, int n0, int k0, int N, int K) {
  int lane = threadIdx.x & 31, half = lane >> 4, l16 = lane & 15;
  int n = n0 + l16;
  bfrag f;
#pragma unroll
  for (int v = 0; v < 16; ++v) {
    int k = k0 + half * 16 + v;
    f[v] = (n < N && k < K) ? (bf16)W[(long)k * ldn + n] : (bf16)0.f;
  }
  return f;
}

template<bool KC, bool NT, typename TB>
DEVI bfrag load_b_any(const TB* __restrict__ Bw, int ldb, int n0, int k0, int N, int K) {
  if constexpr (KC) {
    if constexpr (std::is_same_v<TB, float>) return load_b_kc_f32<NT>(Bw, ldb, n0, k0, N, K);
    else                                     return load_b_kc_bf16(Bw, ldb, n0, k0, N, K);
  } else {
    return load_b_nc_f32(Bw, ldb, n0, k0, N, K);
  }
}

// ---------------------------------------------------------------------------
// Generic GEMM: C[M,N] = A[M,K](f32) @ B  (B f32 or bf16; KC => N-major/K-contig)
// Block = 8 waves; each wave owns 16 output cols; MT 16-row M-tiles per block so
// skinny-M GEMMs read the streamed B operand exactly once. gridDim.z splits K
// (atomicAdd f32 into pre-initialized C).
// ---------------------------------------------------------------------------
template<int MT, bool KC, bool NT, typename TB>
__global__ __launch_bounds__(256) void gemm_wmma_k(
    const float* __restrict__ A, const TB* __restrict__ Bw, float* __restrict__ C,
    int M, int N, int K, int lda, int ldb, int ldc,
    const float* __restrict__ bias, const float* __restrict__ Cadd, int ldcadd,
    int kchunk)
{
  __shared__ __align__(16) bf16 As[MT * 16 * 32];
  int wave = threadIdx.x >> 5;
  int n0 = blockIdx.x * 128 + wave * 16;
  int mbase = blockIdx.y * (MT * 16);
  int kb = blockIdx.z * kchunk;
  int ke = kb + kchunk; if (ke > K) ke = K;

  v8f acc[MT];
#pragma unroll
  for (int mt = 0; mt < MT; ++mt)
#pragma unroll
    for (int i = 0; i < 8; ++i) acc[mt][i] = 0.f;

  for (int k0 = kb; k0 < ke; k0 += 32) {
    for (int i = threadIdx.x; i < MT * 512; i += 256) {
      int r = i >> 5, c = i & 31;
      int m = mbase + r, k = k0 + c;
      As[i] = (m < M && k < K) ? (bf16)A[(long)m * lda + k] : (bf16)0.f;
    }
    __syncthreads();
    if (n0 < N) {
      bfrag bfr = load_b_any<KC, NT>(Bw, ldb, n0, k0, N, K);
#pragma unroll
      for (int mt = 0; mt < MT; ++mt) {
        bfrag afr = load_a_lds_ld(&As[mt * 512], 32);
        acc[mt] = wmma_bf16(afr, bfr, acc[mt]);
      }
    }
    __syncthreads();
  }

  if (n0 >= N) return;
  int lane = threadIdx.x & 31, half = lane >> 4, l16 = lane & 15;
  int n = n0 + l16;
  if (n >= N) return;
  bool split = (gridDim.z > 1);
#pragma unroll
  for (int mt = 0; mt < MT; ++mt) {
#pragma unroll
    for (int r = 0; r < 8; ++r) {
      int m = mbase + mt * 16 + half * 8 + r;
      if (m >= M) continue;
      float v = acc[mt][r];
      if (split) {
        atomicAdd(&C[(long)m * ldc + n], v);
      } else {
        if (bias) v += bias[n];
        if (Cadd) v += Cadd[(long)m * ldcadd + n];
        C[(long)m * ldc + n] = v;
      }
    }
  }
}

template<int MT, bool KC, bool NT, typename TB>
static void launch_gemm(hipStream_t s, const float* A, int lda, const TB* B, int ldb,
                        float* C, int ldc, int M, int N, int K, int zsplit,
                        const float* bias, const float* cadd, int ldcadd) {
  dim3 grid((N + 127) / 128, (M + MT * 16 - 1) / (MT * 16), zsplit);
  int kchunk = (((K + zsplit - 1) / zsplit) + 31) & ~31;
  gemm_wmma_k<MT, KC, NT, TB><<<grid, dim3(256), 0, s>>>(A, B, C, M, N, K, lda, ldb, ldc,
                                                         bias, cadd, ldcadd, kchunk);
}

// ---------------------------------------------------------------------------
// Skinny GEMM for the GRU critical path, M = 32 fixed, A already bf16.
// The 32 x 512 A-slice for this block's K-chunk is staged to LDS with a single
// Tensor Data Mover DMA (tensor_load_to_lds + s_wait_tensorcnt), then 8 waves
// stream the B operand (W_hh bf16 / Wd f32) straight from L2/HBM through WMMA.
// K is split across gridDim.z; partial products atomicAdd into pre-init C.
// ---------------------------------------------------------------------------
template<bool KC, typename TB>
__global__ __launch_bounds__(256) void gemm_m32_tdm_k(
    const bf16* __restrict__ A, const TB* __restrict__ Bw, float* __restrict__ C,
    int N, int K, int lda, int ldb, int ldc, int kchunk)
{
  __shared__ __align__(16) bf16 Asl[32 * 512];
  int wave = threadIdx.x >> 5;
  int n0 = blockIdx.x * 128 + wave * 16;
  int kb = blockIdx.z * kchunk;
  int kc = K - kb; if (kc > kchunk) kc = kchunk;   // == 512 in all launches here

#if HAVE_TDM
  if (threadIdx.x < 32) {
    tdm_load_tile_bf16((unsigned)(size_t)(void*)Asl, A + kb, 32u, (unsigned)kc,
                       (unsigned long long)lda);
    __builtin_amdgcn_s_wait_tensorcnt(0);
  }
#else
  for (int i = threadIdx.x * 8; i < 32 * kc; i += 256 * 8) {
    int r = i / 512, c = i & 511;
    *(bf8*)&Asl[r * 512 + c] = *(const bf8*)&A[(long)r * lda + kb + c];
  }
#endif
  __syncthreads();

  v8f acc0, acc1;
#pragma unroll
  for (int i = 0; i < 8; ++i) { acc0[i] = 0.f; acc1[i] = 0.f; }

  for (int k0 = 0; k0 < kc; k0 += 32) {
    if (n0 < N) {
      bfrag bfr = load_b_any<KC, false>(Bw, ldb, n0, kb + k0, N, K);
      bfrag a0 = load_a_lds_ld(Asl + k0, 512);
      acc0 = wmma_bf16(a0, bfr, acc0);
      bfrag a1 = load_a_lds_ld(Asl + 16 * 512 + k0, 512);
      acc1 = wmma_bf16(a1, bfr, acc1);
    }
  }

  if (n0 >= N) return;
  int lane = threadIdx.x & 31, half = lane >> 4, l16 = lane & 15;
  int n = n0 + l16;
  bool split = (gridDim.z > 1);
#pragma unroll
  for (int r = 0; r < 8; ++r) {
    int m = half * 8 + r;
    if (split) {
      atomicAdd(&C[(long)m * ldc + n], acc0[r]);
      atomicAdd(&C[(long)(m + 16) * ldc + n], acc1[r]);
    } else {
      C[(long)m * ldc + n] = acc0[r];
      C[(long)(m + 16) * ldc + n] = acc1[r];
    }
  }
}

template<bool KC, typename TB>
static void launch_gemm_m32(hipStream_t s, const bf16* A, int lda, const TB* B, int ldb,
                            float* C, int ldc, int N, int K) {
  const int kchunk = 512;
  dim3 grid((N + 127) / 128, 1, (K + kchunk - 1) / kchunk);
  gemm_m32_tdm_k<KC, TB><<<grid, dim3(256), 0, s>>>(A, B, C, N, K, lda, ldb, ldc, kchunk);
}

// ---------------------------------------------------------------------------
// Batched attention pieces: logits = Q K^T * scale and O = P V, per (b, head).
// ---------------------------------------------------------------------------
__global__ __launch_bounds__(256) void k_attn_logits(
    const float* __restrict__ Q, const float* __restrict__ Kc, float* __restrict__ L,
    int nk, float scale)
{
  int bh = blockIdx.x, b = bh >> 3, h = bh & 7;
  int wave = threadIdx.x >> 5;
  int n0 = wave * 16;
  if (n0 >= nk) return;
  const float* Qb = Q  + (size_t)b * 128 * 512 + h * 64;
  const float* Kb = Kc + (size_t)b * nk  * 512 + h * 64;
  v8f acc[8];
#pragma unroll
  for (int mi = 0; mi < 8; ++mi)
#pragma unroll
    for (int i = 0; i < 8; ++i) acc[mi][i] = 0.f;
  for (int kd = 0; kd < 64; kd += 32) {
    bfrag bfr = load_b_kc_f32<false>(Kb, 512, n0, kd, nk, 64);
#pragma unroll
    for (int mi = 0; mi < 8; ++mi) {
      bfrag afr = load_a_g(Qb, 512, mi * 16, kd, 128, 64);
      acc[mi] = wmma_bf16(afr, bfr, acc[mi]);
    }
  }
  int lane = threadIdx.x & 31, half = lane >> 4, l16 = lane & 15;
  int n = n0 + l16;
  float* Lb = L + (size_t)bh * 128 * nk;
#pragma unroll
  for (int mi = 0; mi < 8; ++mi)
#pragma unroll
    for (int r = 0; r < 8; ++r) {
      int m = mi * 16 + half * 8 + r;
      Lb[(size_t)m * nk + n] = acc[mi][r] * scale;
    }
}

__global__ __launch_bounds__(256) void k_attn_av(
    const float* __restrict__ P, const float* __restrict__ V, float* __restrict__ O, int nk)
{
  int bh = blockIdx.x, b = bh >> 3, h = bh & 7;
  int wave = threadIdx.x >> 5;
  int n0 = wave * 16;
  if (n0 >= 64) return;
  const float* Pb = P + (size_t)bh * 128 * nk;
  const float* Vb = V + (size_t)b * nk * 512 + h * 64;
  v8f acc[8];
#pragma unroll
  for (int mi = 0; mi < 8; ++mi)
#pragma unroll
    for (int i = 0; i < 8; ++i) acc[mi][i] = 0.f;
  for (int k0 = 0; k0 < nk; k0 += 32) {
    bfrag bfr = load_b_nc_f32(Vb, 512, n0, k0, 64, nk);
#pragma unroll
    for (int mi = 0; mi < 8; ++mi) {
      bfrag afr = load_a_g(Pb, nk, mi * 16, k0, 128, nk);
      acc[mi] = wmma_bf16(afr, bfr, acc[mi]);
    }
  }
  int lane = threadIdx.x & 31, half = lane >> 4, l16 = lane & 15;
  int n = n0 + l16;
#pragma unroll
  for (int mi = 0; mi < 8; ++mi)
#pragma unroll
    for (int r = 0; r < 8; ++r) {
      int m = mi * 16 + half * 8 + r;
      O[((size_t)b * 128 + m) * 512 + h * 64 + n] = acc[mi][r];
    }
}

// ---------------------------------------------------------------------------
// Row-wise softmax / LayerNorm (one wave32 per row, shuffle reductions)
// ---------------------------------------------------------------------------
__global__ __launch_bounds__(256) void k_softmax(float* __restrict__ L, int W, int rows) {
  int row = blockIdx.x * 8 + (threadIdx.x >> 5);
  if (row >= rows) return;
  int lane = threadIdx.x & 31;
  float* x = L + (size_t)row * W;
  float mx = -1e30f;
  for (int c = lane; c < W; c += 32) mx = fmaxf(mx, x[c]);
  for (int o = 16; o; o >>= 1) mx = fmaxf(mx, __shfl_xor(mx, o));
  float s = 0.f;
  for (int c = lane; c < W; c += 32) { float e = __expf(x[c] - mx); x[c] = e; s += e; }
  for (int o = 16; o; o >>= 1) s += __shfl_xor(s, o);
  float inv = 1.f / s;
  for (int c = lane; c < W; c += 32) x[c] *= inv;
}

__global__ __launch_bounds__(256) void k_ln(
    const float* __restrict__ X, float* __restrict__ Y,
    const float* __restrict__ g, const float* __restrict__ bta, int D, int rows)
{
  int row = blockIdx.x * 8 + (threadIdx.x >> 5);
  if (row >= rows) return;
  int lane = threadIdx.x & 31;
  const float* x = X + (size_t)row * D;
  float s = 0.f, s2 = 0.f;
  for (int c = lane; c < D; c += 32) { float v = x[c]; s += v; s2 += v * v; }
  for (int o = 16; o; o >>= 1) { s += __shfl_xor(s, o); s2 += __shfl_xor(s2, o); }
  float m = s / (float)D;
  float var = s2 / (float)D - m * m;
  float inv = rsqrtf(var + 1e-5f);
  float* y = Y + (size_t)row * D;
  for (int c = lane; c < D; c += 32) y[c] = (x[c] - m) * inv * g[c] + bta[c];
}

// ---------------------------------------------------------------------------
// Small elementwise / data-movement kernels
// ---------------------------------------------------------------------------
__global__ void k_build_ldr(const float* __restrict__ lh, const float* __restrict__ ld,
                            float* __restrict__ out) {
  int i = blockIdx.x * blockDim.x + threadIdx.x;
  if (i >= 32 * 128 * 32) return;
  int c = i & 31, t = (i >> 5) & 127, b = i >> 12;
  const float* src = (t < 64) ? ld : lh;            // first 64 tokens: dense_range, then hdmap
  int p = t & 63;
  out[i] = src[((b * 32 + c) << 6) + p];            // [B, D=32, 64] -> [B, 64, 32]
}

__global__ void k_gather_act(const float* __restrict__ action, float* __restrict__ AP) {
  int i = blockIdx.x * blockDim.x + threadIdx.x;
  if (i >= 224 * 128) return;
  int j = i & 127, r = i >> 7;
  int t = r >> 5, b = r & 31;                       // row = t*32 + b
  AP[i] = action[(b * 8 + t) * 128 + j];
}

__global__ void k_f32_to_bf16(const float* __restrict__ src, bf16* __restrict__ dst, long n) {
  long i = (long)blockIdx.x * blockDim.x + threadIdx.x;
  long stride = (long)gridDim.x * blockDim.x;
  for (; i < n; i += stride)
    dst[i] = (bf16)__builtin_nontemporal_load(src + i);   // NT read: keep L2 for bf16 copy
}

__global__ void k_bias_rows(float* __restrict__ C, const float* __restrict__ bias,
                            int rows, int N) {
  int i = blockIdx.x * blockDim.x + threadIdx.x;
  if (i >= rows * N) return;
  C[i] = bias[i % N];
}

__global__ void k_gru(const float* __restrict__ GI, int t,
                      const float* __restrict__ GH,
                      const float* __restrict__ b_ih, const float* __restrict__ b_hh,
                      const float* __restrict__ xc, float* __restrict__ xn,
                      bf16* __restrict__ xnb)
{
  int i = blockIdx.x * blockDim.x + threadIdx.x;
  if (i >= 32 * 4096) return;
  int b = i >> 12, e = i & 4095;
  const float* gi = GI + (size_t)(t * 32 + b) * 12288;
  const float* gh = GH + (size_t)b * 12288;
  float ir  = gi[e]        + b_ih[e];
  float iz  = gi[4096 + e] + b_ih[4096 + e];
  float in_ = gi[8192 + e] + b_ih[8192 + e];
  float hr  = gh[e]        + b_hh[e];
  float hz  = gh[4096 + e] + b_hh[4096 + e];
  float hn  = gh[8192 + e] + b_hh[8192 + e];
  float r = 1.f / (1.f + __expf(-(ir + hr)));
  float z = 1.f / (1.f + __expf(-(iz + hz)));
  float n = tanhf(in_ + r * hn);
  float x = (1.f - z) * n + z * xc[i];
  xn[i] = x;
  xnb[i] = (bf16)x;      // bf16 mirror feeds the TDM-staged GEMMs
}

// ---------------------------------------------------------------------------
// Orchestration
// ---------------------------------------------------------------------------
extern "C" void kernel_launch(void* const* d_in, const int* in_sizes, int n_in,
                              void* d_out, int out_size, void* d_ws, size_t ws_size,
                              hipStream_t stream) {
  (void)in_sizes; (void)n_in; (void)out_size; (void)ws_size;
  const float* lh     = (const float*)d_in[0];
  const float* box    = (const float*)d_in[1];
  const float* act    = (const float*)d_in[2];
  const float* ldri   = (const float*)d_in[3];
  const float* r2b_Wq = (const float*)d_in[4];
  const float* r2b_Wk = (const float*)d_in[5];
  const float* r2b_Wv = (const float*)d_in[6];
  const float* r2b_Wo = (const float*)d_in[7];
  const float* r2b_bo = (const float*)d_in[8];
  const float* g_r2b  = (const float*)d_in[9];
  const float* b_r2b  = (const float*)d_in[10];
  const float* a2_Wq  = (const float*)d_in[11];
  const float* a2_Wk  = (const float*)d_in[12];
  const float* a2_Wv  = (const float*)d_in[13];
  const float* a2_Wo  = (const float*)d_in[14];
  const float* a2_bo  = (const float*)d_in[15];
  const float* g1     = (const float*)d_in[16];
  const float* b1     = (const float*)d_in[17];
  const float* g2     = (const float*)d_in[18];
  const float* b2     = (const float*)d_in[19];
  const float* a3_Wq  = (const float*)d_in[20];
  const float* a3_Wk  = (const float*)d_in[21];
  const float* a3_Wv  = (const float*)d_in[22];
  const float* a3_Wo  = (const float*)d_in[23];
  const float* a3_bo  = (const float*)d_in[24];
  // d_in[25..33]: g4,b4,a4_*, g51,b51 -- mathematically dead (length-1 attention)
  const float* g52    = (const float*)d_in[34];
  const float* b52    = (const float*)d_in[35];
  const float* a5_Wv  = (const float*)d_in[38];
  const float* a5_Wo  = (const float*)d_in[39];
  const float* a5_bo  = (const float*)d_in[40];
  const float* W_ih   = (const float*)d_in[41];
  const float* W_hh   = (const float*)d_in[42];
  const float* b_ih   = (const float*)d_in[43];
  const float* b_hh   = (const float*)d_in[44];
  const float* Wd     = (const float*)d_in[45];
  const float* bd     = (const float*)d_in[46];
  float* out = (float*)d_out;

  // ---- workspace layout ----
  bf16* WHH = (bf16*)d_ws;                                   // 12288*4096 bf16 (100.7 MB)
  char* p = (char*)d_ws + (size_t)12288 * 4096 * 2;
  bf16* XAB = (bf16*)p; p += (size_t)32 * 4096 * 2;          // bf16 GRU state mirrors
  bf16* XBB = (bf16*)p; p += (size_t)32 * 4096 * 2;
  float* F = (float*)p;
  size_t off = 0;
  auto alloc = [&](size_t n) { float* q = F + off; off += n; return q; };
  float* LDR   = alloc(131072);            // [B*128, 32]
  float* QB    = alloc(2097152);           // [B*128, 512]
  float* KB    = alloc(2097152);
  float* VB    = alloc(2097152);
  float* LG    = alloc(4194304);           // [B*H, 128, <=128]
  float* OB    = alloc(2097152);
  float* R2B   = alloc(131072);
  float* R2BN  = alloc(131072);
  float* R2B2  = alloc(131072);
  float* LN1   = alloc(131072);
  float* LN2   = alloc(131072);
  float* X     = alloc(131072);            // [B, 4096]
  float* AP    = alloc(224 * 128);
  float* LNA   = alloc(224 * 128);
  float* S1    = alloc(224 * 512);
  float* S     = alloc((size_t)224 * 4096);
  float* GI    = alloc((size_t)224 * 12288);
  float* APROJ = alloc(224 * 1024);
  float* GH    = alloc(32 * 12288);
  float* XA    = alloc(131072);
  float* XB    = alloc(131072);

  // W_hh -> bf16 once (reused 7x; 100 MB bf16 is L2-resident afterwards)
  k_f32_to_bf16<<<2048, 256, 0, stream>>>(W_hh, WHH, (long)12288 * 4096);

  // ---- prelude: ldr = [dense_range ; hdmap] tokens ----
  k_build_ldr<<<(32 * 128 * 32 + 255) / 256, 256, 0, stream>>>(lh, ldri, LDR);

  // attention 1: r2b = attn(ldr, boxes)
  launch_gemm<1, false, false, float>(stream, LDR, 32, r2b_Wq, 512, QB, 512, 4096, 512, 32, 1, nullptr, nullptr, 0);
  launch_gemm<1, false, false, float>(stream, box, 16, r2b_Wk, 512, KB, 512, 2048, 512, 16, 1, nullptr, nullptr, 0);
  launch_gemm<1, false, false, float>(stream, box, 16, r2b_Wv, 512, VB, 512, 2048, 512, 16, 1, nullptr, nullptr, 0);
  k_attn_logits<<<256, 256, 0, stream>>>(QB, KB, LG, 64, 0.125f);
  k_softmax<<<(32768 + 7) / 8, 256, 0, stream>>>(LG, 64, 32768);
  k_attn_av<<<256, 256, 0, stream>>>(LG, VB, OB, 64);
  launch_gemm<1, false, false, float>(stream, OB, 512, r2b_Wo, 32, R2B, 32, 4096, 32, 512, 1, r2b_bo, nullptr, 0);
  k_ln<<<512, 256, 0, stream>>>(R2B, R2BN, g_r2b, b_r2b, 32, 4096);

  // attention 2: self-attn on r2b_n
  launch_gemm<1, false, false, float>(stream, R2BN, 32, a2_Wq, 512, QB, 512, 4096, 512, 32, 1, nullptr, nullptr, 0);
  launch_gemm<1, false, false, float>(stream, R2BN, 32, a2_Wk, 512, KB, 512, 4096, 512, 32, 1, nullptr, nullptr, 0);
  launch_gemm<1, false, false, float>(stream, R2BN, 32, a2_Wv, 512, VB, 512, 4096, 512, 32, 1, nullptr, nullptr, 0);
  k_attn_logits<<<256, 256, 0, stream>>>(QB, KB, LG, 128, 0.125f);
  k_softmax<<<(32768 + 7) / 8, 256, 0, stream>>>(LG, 128, 32768);
  k_attn_av<<<256, 256, 0, stream>>>(LG, VB, OB, 128);
  launch_gemm<1, false, false, float>(stream, OB, 512, a2_Wo, 32, R2B2, 32, 4096, 32, 512, 1, a2_bo, nullptr, 0);

  // x = ldr + attn(ln(ldr), ln(r2b2))
  k_ln<<<512, 256, 0, stream>>>(LDR, LN1, g1, b1, 32, 4096);
  k_ln<<<512, 256, 0, stream>>>(R2B2, LN2, g2, b2, 32, 4096);
  launch_gemm<1, false, false, float>(stream, LN1, 32, a3_Wq, 512, QB, 512, 4096, 512, 32, 1, nullptr, nullptr, 0);
  launch_gemm<1, false, false, float>(stream, LN2, 32, a3_Wk, 512, KB, 512, 4096, 512, 32, 1, nullptr, nullptr, 0);
  launch_gemm<1, false, false, float>(stream, LN2, 32, a3_Wv, 512, VB, 512, 4096, 512, 32, 1, nullptr, nullptr, 0);
  k_attn_logits<<<256, 256, 0, stream>>>(QB, KB, LG, 128, 0.125f);
  k_softmax<<<(32768 + 7) / 8, 256, 0, stream>>>(LG, 128, 32768);
  k_attn_av<<<256, 256, 0, stream>>>(LG, VB, OB, 128);
  launch_gemm<1, false, false, float>(stream, OB, 512, a3_Wo, 32, X, 32, 4096, 32, 512, 1, a3_bo, LDR, 32);

  // ---- batched step inputs: s_t depends only on action[:,t] (len-1 attention collapses) ----
  k_gather_act<<<(224 * 128 + 255) / 256, 256, 0, stream>>>(act, AP);
  k_ln<<<28, 256, 0, stream>>>(AP, LNA, g52, b52, 128, 224);
  launch_gemm<1, false, false, float>(stream, LNA, 128, a5_Wv, 512, S1, 512, 224, 512, 128, 1, nullptr, nullptr, 0);
  launch_gemm<1, false, false, float>(stream, S1, 512, a5_Wo, 4096, S, 4096, 224, 4096, 512, 1, a5_bo, nullptr, 0);
  // GI = S @ W_ih^T -- single pass over W_ih (201 MB, read exactly once, non-temporal)
  hipMemsetAsync(GI, 0, (size_t)224 * 12288 * 4, stream);
  launch_gemm<14, true, true, float>(stream, S, 4096, W_ih, 4096, GI, 12288, 224, 12288, 4096, 4, nullptr, nullptr, 0);
  // APROJ[t] = a_prev_t @ Wd_bot + bd (batched)
  launch_gemm<1, false, false, float>(stream, AP, 128, Wd + (size_t)4096 * 1024, 1024, APROJ, 1024,
                                      224, 1024, 128, 1, bd, nullptr, 0);

  // h0 = x @ Wd_top + bd  (TDM-staged skinny GEMM)
  k_f32_to_bf16<<<128, 256, 0, stream>>>(X, XAB, 131072);
  k_bias_rows<<<(32 * 1024 + 255) / 256, 256, 0, stream>>>(out, bd, 32, 1024);
  launch_gemm_m32<false, float>(stream, XAB, 4096, Wd, 1024, out, 1024, 1024, 4096);

  // ---- sequential GRU: only W_hh (bf16, L2-resident) is re-read per step ----
  hipMemcpyAsync(XA, X, 131072 * 4, hipMemcpyDeviceToDevice, stream);
  float* xc = XA; float* xn = XB;
  bf16* xcb = XAB; bf16* xnb = XBB;
  for (int t = 0; t < 7; ++t) {
    hipMemsetAsync(GH, 0, (size_t)32 * 12288 * 4, stream);
    launch_gemm_m32<true, bf16>(stream, xcb, 4096, WHH, 4096, GH, 12288, 12288, 4096);
    k_gru<<<512, 256, 0, stream>>>(GI, t, GH, b_ih, b_hh, xc, xn, xnb);
    float* outt = out + (size_t)(t + 1) * 32 * 1024;
    hipMemcpyAsync(outt, APROJ + (size_t)t * 32 * 1024, 32 * 1024 * 4,
                   hipMemcpyDeviceToDevice, stream);
    launch_gemm_m32<false, float>(stream, xnb, 4096, Wd, 1024, outt, 1024, 1024, 4096);
    float* tf = xc; xc = xn; xn = tf;
    bf16* tb = xcb; xcb = xnb; xnb = tb;
  }
}
